// LucaGPLMMultiheadAttentionWithSDPA_18081812316334
// MI455X (gfx1250) — matrix operand order repaired and checked by reading, not compiled, not run on hardware
//
#include <hip/hip_runtime.h>
#include <math.h>

// ---------------------------------------------------------------------------
// Types for CDNA5 WMMA (wave32, 16x16x32 bf16 -> f32)
// ---------------------------------------------------------------------------
typedef __attribute__((ext_vector_type(16))) __bf16 v16bf;
typedef __attribute__((ext_vector_type(8)))  __bf16 v8bf;
typedef __attribute__((ext_vector_type(8)))  float  v8f;

union Frag {
    v16bf v;
    v8bf  h[2];
};

static __device__ __forceinline__ unsigned short f32_to_bf16(float f) {
    unsigned int u = __float_as_uint(f);
    unsigned int r = (u + 0x7FFFu + ((u >> 16) & 1u)) >> 16;  // RNE
    return (unsigned short)r;
}

#define WMMA_BF16(A, Bm, C) \
    __builtin_amdgcn_wmma_f32_16x16x32_bf16(false, (A), false, (Bm), (short)0, (C), false, false)

// gfx1250 async memory->LDS copy (ASYNCcnt-tracked).
// IMPORTANT: the LDS pointer is passed through ptrtoint so its address ESCAPES
// into the asm -- otherwise the compiler proves the __shared__ array is never
// stored and folds all ds_loads of it to poison. Low 32 bits of a flat pointer
// into the shared aperture are exactly the LDS byte address (ISA 10.2).
static __device__ __forceinline__ void async_ld_b128(void* lptr, const void* gptr) {
    unsigned ldsOff = (unsigned)(unsigned long long)lptr;
    asm volatile("global_load_async_to_lds_b128 %0, %1, off"
                 :: "v"(ldsOff), "v"(gptr) : "memory");
}

// ---------------------------------------------------------------------------
// fp32 -> bf16 elementwise convert
// ---------------------------------------------------------------------------
__global__ void cvt_bf16_kernel(const float* __restrict__ in,
                                unsigned short* __restrict__ out, int n) {
    int i = blockIdx.x * blockDim.x + threadIdx.x;
    if (i < n) out[i] = f32_to_bf16(in[i]);
}

// ---------------------------------------------------------------------------
// WMMA GEMM: C[M,N] (f32) = A_bf16[M,K] * W_bf16[N,K]^T + bias[N]
// Block tile 128x128, K-step 32. Double-buffered LDS filled with
// global_load_async_to_lds_b128 (ASYNCcnt ping-pong). 8 waves, each computes
// a 32x64 sub-tile = 8 WMMA per K-step.
// ---------------------------------------------------------------------------
#define GBM 128
#define GBN 128
#define GBK 32
#define LPAD 8

__global__ __launch_bounds__(256) void gemm_bf16_kernel(
    const unsigned short* __restrict__ A, const unsigned short* __restrict__ W,
    const float* __restrict__ bias, float* __restrict__ C, int M, int N, int K) {
    constexpr int LDT   = GBK + LPAD;    // 40 halves per tile row (conflict-free)
    constexpr int ATILE = GBM * LDT;     // 5120 halves
    constexpr int BTILE = GBN * LDT;     // 5120 halves
    constexpr int BUF   = ATILE + BTILE; // 10240 halves per ping-pong buffer
    __shared__ unsigned short lds[2 * BUF];  // 40 KB

    const int t    = threadIdx.x;
    const int wave = t >> 5;
    const int lane = t & 31;
    const int hi   = lane >> 4;
    const int ln   = lane & 15;
    const int wm   = wave >> 1;   // 0..3 -> 32 M-rows
    const int wn   = wave & 1;    // 0..1 -> 64 N-cols
    const int bm0  = blockIdx.y * GBM;
    const int bn0  = blockIdx.x * GBN;

    // async fill mapping: 128x32-half tile, 32 B per thread (2 x b128)
    const int frow = t >> 1;
    const int fcol = (t & 1) * 16;
    const unsigned short* gA = A + (size_t)(bm0 + frow) * K + fcol;
    const unsigned short* gW = W + (size_t)(bn0 + frow) * K + fcol;
    unsigned short* const lA = lds + frow * LDT + fcol;
    unsigned short* const lB = lds + ATILE + frow * LDT + fcol;

    auto issue_tile = [&](int buf, int kk) {
        unsigned short* la = lA + buf * BUF;
        unsigned short* lb = lB + buf * BUF;
        async_ld_b128(la,     gA + kk);
        async_ld_b128(la + 8, gA + kk + 8);
        async_ld_b128(lb,     gW + kk);
        async_ld_b128(lb + 8, gW + kk + 8);
    };

    v8f acc[2][4];
    for (int i = 0; i < 2; ++i)
        for (int j = 0; j < 4; ++j) acc[i][j] = (v8f)0.0f;

    auto compute_tile = [&](int buf) {
        const unsigned short* As = lds + buf * BUF;
        const unsigned short* Bs = As + ATILE;
        Frag fa[2], fb[4];
        for (int mt = 0; mt < 2; ++mt) {
            int r = wm * 32 + mt * 16 + ln;
            // A-layout (16x32 bf16): lane holds K chunks {8*hi..} and {16+8*hi..}
            fa[mt].h[0] = *(const v8bf*)(As + r * LDT + hi * 8);
            fa[mt].h[1] = *(const v8bf*)(As + r * LDT + 16 + hi * 8);
        }
        for (int nt = 0; nt < 4; ++nt) {
            int c = wn * 64 + nt * 16 + ln;
            // B-layout (32x16 bf16): lane = column, K chunks {8*hi..} and {16+8*hi..}
            fb[nt].h[0] = *(const v8bf*)(Bs + c * LDT + hi * 8);
            fb[nt].h[1] = *(const v8bf*)(Bs + c * LDT + 16 + hi * 8);
        }
        for (int mt = 0; mt < 2; ++mt)
            for (int nt = 0; nt < 4; ++nt)
                acc[mt][nt] = WMMA_BF16(fa[mt].v, fb[nt].v, acc[mt][nt]);
    };

    const int nk = K / GBK;
    issue_tile(0, 0);
    for (int j = 0; j < nk - 1; ++j) {
        issue_tile((j + 1) & 1, (j + 1) * GBK);
        // 4 newly issued outstanding; previous tile's 4 (in-order) have landed
        asm volatile("s_wait_asynccnt 0x4" ::: "memory");
        __syncthreads();
        compute_tile(j & 1);
        __syncthreads();  // protect buffer until all waves done reading
    }
    asm volatile("s_wait_asynccnt 0x0" ::: "memory");
    __syncthreads();
    compute_tile((nk - 1) & 1);

    // epilogue: C-layout: vgpr v <-> row hi*8+v, lane ln <-> col
    for (int mt = 0; mt < 2; ++mt) {
        for (int nt = 0; nt < 4; ++nt) {
            int col  = bn0 + wn * 64 + nt * 16 + ln;
            float bv = bias[col];
            for (int v = 0; v < 8; ++v) {
                int row = bm0 + wm * 32 + mt * 16 + hi * 8 + v;
                C[(size_t)row * N + col] = acc[mt][nt][v] + bv;
            }
        }
    }
}

// ---------------------------------------------------------------------------
// RoPE + pack to bf16 [B,H,S,hd]; Q additionally folds the 1/sqrt(hd) scale.
// ---------------------------------------------------------------------------
__global__ void rope_pack_kernel(const float* __restrict__ F,
                                 unsigned short* __restrict__ out,
                                 int S, int B, int H, float scale) {
    int idx = blockIdx.x * blockDim.x + threadIdx.x;  // [(b*H+h)*S+s]*64 + d
    int d   = idx & 63;
    int tmp = idx >> 6;
    int s   = tmp % S;
    int bh  = tmp / S;
    int h   = bh % H;
    int b   = bh / H;
    int E   = H * 64;
    size_t rowb = (size_t)(s * B + b) * E + h * 64;
    float x  = F[rowb + d];
    float xr = F[rowb + (d < 32 ? d + 32 : d - 32)];
    float rot = (d < 32) ? -xr : xr;
    int   j   = d & 31;
    float invf = expf(-(float)j * (9.2103403719761836f / 32.0f));  // 10000^(-j/32)
    float th   = (float)s * invf;
    float o    = (x * cosf(th) + rot * sinf(th)) * scale;
    out[idx] = f32_to_bf16(o);
}

// V packed TRANSPOSED: bf16 [B,H,hd,S] so P@V B-fragments are contiguous.
__global__ void vpack_kernel(const float* __restrict__ F,
                             unsigned short* __restrict__ out,
                             int S, int B, int H) {
    int idx = blockIdx.x * blockDim.x + threadIdx.x;  // [(b*H+h)*64+d]*S + s
    int s   = idx % S;
    int tmp = idx / S;
    int d   = tmp & 63;
    int bh  = tmp >> 6;
    int h   = bh % H;
    int b   = bh / H;
    int E   = H * 64;
    out[idx] = f32_to_bf16(F[(size_t)(s * B + b) * E + h * 64 + d]);
}

// ---------------------------------------------------------------------------
// Flash attention. grid = (S/128, B*H), 8 waves/block, each wave owns 16 q rows.
// Q,K: bf16 [B,H,S,64] (RoPE'd, Q pre-scaled). V: bf16 [B,H,64,S] (transposed).
// Online softmax; P goes through a wave-private LDS tile for C->A relayout.
// ---------------------------------------------------------------------------
__global__ __launch_bounds__(256) void attn_kernel(
    const unsigned short* __restrict__ Q, const unsigned short* __restrict__ K,
    const unsigned short* __restrict__ Vt, unsigned short* __restrict__ Ctx,
    int S, int B, int H) {
    __shared__ unsigned short Pb[8][16 * 32];

    const int t    = threadIdx.x;
    const int wave = t >> 5;
    const int lane = t & 31;
    const int hi   = lane >> 4;
    const int ln   = lane & 15;
    const int bh   = blockIdx.y;
    const int h    = bh % H;
    const int b    = bh / H;
    const int q0   = blockIdx.x * 128 + wave * 16;
    const int E    = H * 64;
    const size_t qkbase = (size_t)bh * S * 64;

    // Q A-fragments (held in registers for the whole kernel): 16 rows x 64 hd
    Frag qf[2];
    {
        const unsigned short* p = Q + qkbase + (size_t)(q0 + ln) * 64;
        for (int c = 0; c < 2; ++c) {
            qf[c].h[0] = *(const v8bf*)(p + c * 32 + hi * 8);
            qf[c].h[1] = *(const v8bf*)(p + c * 32 + 16 + hi * 8);
        }
    }

    float mrun[8], lrun[8];
    v8f acc[4];
    for (int v = 0; v < 8; ++v) { mrun[v] = -INFINITY; lrun[v] = 0.0f; }
    for (int j = 0; j < 4; ++j) acc[j] = (v8f)0.0f;

    for (int kb = 0; kb < S; kb += 32) {
        // ---- scores: two 16x16 tiles over 32 keys, contract over hd=64 ----
        v8f s01[2];
        s01[0] = (v8f)0.0f;
        s01[1] = (v8f)0.0f;
        for (int nt = 0; nt < 2; ++nt) {
            const unsigned short* kp = K + qkbase + (size_t)(kb + nt * 16 + ln) * 64;
            for (int c = 0; c < 2; ++c) {
                Frag kf;
                kf.h[0] = *(const v8bf*)(kp + c * 32 + hi * 8);
                kf.h[1] = *(const v8bf*)(kp + c * 32 + 16 + hi * 8);
                s01[nt] = WMMA_BF16(qf[c].v, kf.v, s01[nt]);
            }
        }

        // ---- online softmax (row stats; rows live in 16-lane halves) ----
        for (int v = 0; v < 8; ++v) {
            float tm = fmaxf(s01[0][v], s01[1][v]);
            tm = fmaxf(tm, __shfl_xor(tm, 1));
            tm = fmaxf(tm, __shfl_xor(tm, 2));
            tm = fmaxf(tm, __shfl_xor(tm, 4));
            tm = fmaxf(tm, __shfl_xor(tm, 8));
            float nm = fmaxf(mrun[v], tm);
            float sc = __expf(mrun[v] - nm);
            mrun[v] = nm;
            float p0 = __expf(s01[0][v] - nm);
            float p1 = __expf(s01[1][v] - nm);
            float rs = p0 + p1;
            rs += __shfl_xor(rs, 1);
            rs += __shfl_xor(rs, 2);
            rs += __shfl_xor(rs, 4);
            rs += __shfl_xor(rs, 8);
            lrun[v] = lrun[v] * sc + rs;
            acc[0][v] *= sc; acc[1][v] *= sc; acc[2][v] *= sc; acc[3][v] *= sc;
            int row = hi * 8 + v;
            Pb[wave][row * 32 + ln]      = f32_to_bf16(p0);
            Pb[wave][row * 32 + 16 + ln] = f32_to_bf16(p1);
        }
        asm volatile("s_wait_dscnt 0" ::: "memory");  // wave-local LDS visibility

        // ---- P (A-frag, 16x32) and P @ V over hd (4 N-tiles) ----
        Frag pf;
        pf.h[0] = *(const v8bf*)&Pb[wave][ln * 32 + hi * 8];
        pf.h[1] = *(const v8bf*)&Pb[wave][ln * 32 + 16 + hi * 8];
        for (int j = 0; j < 4; ++j) {
            Frag vf;
            const unsigned short* vp =
                Vt + (size_t)(bh * 64 + j * 16 + ln) * S + kb + hi * 8;
            vf.h[0] = *(const v8bf*)(vp);
            vf.h[1] = *(const v8bf*)(vp + 16);
            acc[j] = WMMA_BF16(pf.v, vf.v, acc[j]);
        }
    }

    // ---- normalize and write context bf16 as [S,B,E] rows ----
    for (int v = 0; v < 8; ++v) {
        float inv = 1.0f / lrun[v];
        int s = q0 + hi * 8 + v;
        size_t rowb = (size_t)(s * B + b) * E + h * 64;
        for (int j = 0; j < 4; ++j)
            Ctx[rowb + j * 16 + ln] = f32_to_bf16(acc[j][v] * inv);
    }
}

// ---------------------------------------------------------------------------
// Host-side launcher
// ---------------------------------------------------------------------------
extern "C" void kernel_launch(void* const* d_in, const int* in_sizes, int n_in,
                              void* d_out, int out_size, void* d_ws, size_t ws_size,
                              hipStream_t stream) {
    (void)in_sizes; (void)n_in; (void)out_size; (void)ws_size;
    const float* query = (const float*)d_in[0];
    const float* wq = (const float*)d_in[1];
    const float* bq = (const float*)d_in[2];
    const float* wk = (const float*)d_in[3];
    const float* bk = (const float*)d_in[4];
    const float* wv = (const float*)d_in[5];
    const float* bv = (const float*)d_in[6];
    const float* wo = (const float*)d_in[7];
    const float* bo = (const float*)d_in[8];

    const int S = 2048, B = 2, E = 1024, H = 16;
    const int M = S * B;  // 4096

    char* ws = (char*)d_ws;
    const size_t MB = 1u << 20;
    unsigned short* Xb  = (unsigned short*)(ws + 0 * MB);   // 8 MB  bf16 activations
    unsigned short* Wqb = (unsigned short*)(ws + 8 * MB);   // 2 MB each
    unsigned short* Wkb = (unsigned short*)(ws + 10 * MB);
    unsigned short* Wvb = (unsigned short*)(ws + 12 * MB);
    unsigned short* Wob = (unsigned short*)(ws + 14 * MB);
    float*          Ff  = (float*)(ws + 16 * MB);           // 16 MB fp32 GEMM scratch
    unsigned short* Qb  = (unsigned short*)(ws + 32 * MB);  // 8 MB [B,H,S,64]
    unsigned short* Kb  = (unsigned short*)(ws + 40 * MB);  // 8 MB [B,H,S,64]
    unsigned short* Vt  = (unsigned short*)(ws + 48 * MB);  // 8 MB [B,H,64,S]
    unsigned short* Ctx = (unsigned short*)(ws + 56 * MB);  // 8 MB bf16 context

    cvt_bf16_kernel<<<(M * E) / 256, 256, 0, stream>>>(query, Xb, M * E);
    cvt_bf16_kernel<<<(E * E) / 256, 256, 0, stream>>>(wq, Wqb, E * E);
    cvt_bf16_kernel<<<(E * E) / 256, 256, 0, stream>>>(wk, Wkb, E * E);
    cvt_bf16_kernel<<<(E * E) / 256, 256, 0, stream>>>(wv, Wvb, E * E);
    cvt_bf16_kernel<<<(E * E) / 256, 256, 0, stream>>>(wo, Wob, E * E);

    dim3 gg(E / GBN, M / GBM);
    int  pk = (B * H * S * 64) / 256;

    gemm_bf16_kernel<<<gg, 256, 0, stream>>>(Xb, Wqb, bq, Ff, M, E, E);
    rope_pack_kernel<<<pk, 256, 0, stream>>>(Ff, Qb, S, B, H, 0.125f);  // * 1/sqrt(64)

    gemm_bf16_kernel<<<gg, 256, 0, stream>>>(Xb, Wkb, bk, Ff, M, E, E);
    rope_pack_kernel<<<pk, 256, 0, stream>>>(Ff, Kb, S, B, H, 1.0f);

    gemm_bf16_kernel<<<gg, 256, 0, stream>>>(Xb, Wvb, bv, Ff, M, E, E);
    vpack_kernel<<<pk, 256, 0, stream>>>(Ff, Vt, S, B, H);

    dim3 ga(S / 128, B * H);
    attn_kernel<<<ga, 256, 0, stream>>>(Qb, Kb, Vt, Ctx, S, B, H);

    gemm_bf16_kernel<<<gg, 256, 0, stream>>>(Ctx, Wob, bo, (float*)d_out, M, E, E);
}
